// CorticalColumn_26336739459346
// MI455X (gfx1250) — matrix operand order repaired and checked by reading, not compiled, not run on hardware
//
#include <hip/hip_runtime.h>
#include <hip/hip_bf16.h>

// ---------------------------------------------------------------------------
// CorticalColumn forward for MI455X (gfx1250, wave32, WMMA, async-to-LDS).
//
//   h = x @ W_in^T + b_in
//   3x { LN -> GELU -> top-k mask ; h = h @ (ff_w*ff_mask)^T + ff_b }
//   LN -> GELU -> top-k mask ; out = h @ W_out^T + b_out
//
// GEMMs use v_wmma_f32_16x16x32_bf16 with split-bf16 (hi+lo) operands and f32
// accumulation (3 WMMAs / product) for ~fp32 accuracy on the fast bf16 pipe.
// Activations travel between kernels pre-split into bf16 hi/lo (same bytes as
// f32), so the GEMM inner loop is pure GLOBAL_LOAD_ASYNC_TO_LDS_B128 copies
// (ASYNCcnt-tracked, double buffered) + ds_load_b128 + WMMA.
//
// Workspace (~210 MB): [Act hi | Act lo | split weights hi/lo x5]
// ---------------------------------------------------------------------------

typedef __bf16 bf16_t;
typedef __attribute__((ext_vector_type(16))) __bf16 v16bf;
typedef __attribute__((ext_vector_type(8)))  __bf16 v8bf;
typedef __attribute__((ext_vector_type(8)))  float  v8f;

#define TILE_M 128
#define TILE_N 128
#define TILE_K 32
#define LDSA   40   // padded K-stride (bf16): 80B rows -> 16B aligned, bank-clean

#if __has_builtin(__builtin_amdgcn_s_wait_asynccnt)
#define WAIT_ASYNC(n) __builtin_amdgcn_s_wait_asynccnt((unsigned short)(n))
#else
#define WAIT_ASYNC(n) asm volatile("s_wait_asynccnt %0" ::"i"(n) : "memory")
#endif

// One 16B async copy: LDS[lds_off] = *(global base + goff).  GVS addressing:
// SGPR64 base + VGPR32 byte offset; VDST VGPR carries the LDS byte offset.
static __device__ __forceinline__ void async_cp16(unsigned lds_off,
                                                  const void* base,
                                                  unsigned goff) {
  asm volatile("global_load_async_to_lds_b128 %0, %1, %2 offset:0"
               :: "v"(lds_off), "v"(goff), "s"(base)
               : "memory");
}

static __device__ __forceinline__ v16bf cat8(v8bf a, v8bf b) {
  return __builtin_shufflevector(a, b, 0, 1, 2, 3, 4, 5, 6, 7,
                                       8, 9, 10, 11, 12, 13, 14, 15);
}

// ---------------- split f32 -> bf16 hi/lo (optionally masked) ---------------
__global__ __launch_bounds__(256) void cc_prep_split(
    const float* __restrict__ w, const float* __restrict__ mask,
    bf16_t* __restrict__ whi, bf16_t* __restrict__ wlo, int n) {
  for (int i = blockIdx.x * blockDim.x + threadIdx.x; i < n;
       i += gridDim.x * blockDim.x) {
    float v = w[i];
    if (mask) v *= mask[i];
    bf16_t hi = (bf16_t)v;
    whi[i] = hi;
    wlo[i] = (bf16_t)(v - (float)hi);
  }
}

// ---------------- split-bf16 GEMM:  C = A @ W^T + bias ----------------------
// Ahi/Alo:[M,K] bf16, Whi/Wlo:[N,K] bf16 (both K-major), C:[M,N] f32.
__global__ __launch_bounds__(256) void cc_gemm_wmma(
    const bf16_t* __restrict__ Ahi, const bf16_t* __restrict__ Alo,
    const bf16_t* __restrict__ Whi, const bf16_t* __restrict__ Wlo,
    const float* __restrict__ bias, float* __restrict__ C,
    int M, int N, int K) {
  __shared__ bf16_t smem[2][4][TILE_M * LDSA];  // [buf][Ahi,Alo,Bhi,Blo]

  const int t = threadIdx.x;
  const int wave = t >> 5;
  const int lane = t & 31;
  const int hw = lane >> 4;
  const int lr = lane & 15;
  const int wr = wave >> 2;   // 0..1 -> 64-row span
  const int wc = wave & 3;    // 0..3 -> 32-col span
  const int m0 = blockIdx.x * TILE_M;
  const int n0 = blockIdx.y * TILE_N;

  // Each thread copies two 16B chunks per array per k-tile (identical row/kc
  // mapping for A and B slabs: 128 rows x 32 bf16).
  unsigned ldsoff[2], goff[2];
#pragma unroll
  for (int p = 0; p < 2; ++p) {
    int q = p * 256 + t;          // 0..511
    int row = q >> 2;             // 0..127
    int kc = (q & 3) * 8;         // bf16 elems
    ldsoff[p] = (unsigned)(row * LDSA + kc) * 2u;
    goff[p] = (unsigned)(row * K + kc) * 2u;
  }
  const unsigned lds_base = (unsigned)(size_t)(&smem[0][0][0]);
  const unsigned arrB = (unsigned)(TILE_M * LDSA) * 2u;  // bytes per array
  const bf16_t* baseAhi = Ahi + (size_t)m0 * K;
  const bf16_t* baseAlo = Alo + (size_t)m0 * K;
  const bf16_t* baseBhi = Whi + (size_t)n0 * K;
  const bf16_t* baseBlo = Wlo + (size_t)n0 * K;

  auto issue_tile = [&](int k0, int buf) {
    const unsigned kb = (unsigned)k0 * 2u;
    const unsigned b0 = lds_base + (unsigned)buf * 4u * arrB;
#pragma unroll
    for (int p = 0; p < 2; ++p) {
      async_cp16(b0 + 0u * arrB + ldsoff[p], baseAhi, goff[p] + kb);
      async_cp16(b0 + 1u * arrB + ldsoff[p], baseAlo, goff[p] + kb);
      async_cp16(b0 + 2u * arrB + ldsoff[p], baseBhi, goff[p] + kb);
      async_cp16(b0 + 3u * arrB + ldsoff[p], baseBlo, goff[p] + kb);
    }
  };

  v8f acc[4][2];
  const v8f vzero = {0.f, 0.f, 0.f, 0.f, 0.f, 0.f, 0.f, 0.f};
#pragma unroll
  for (int mt = 0; mt < 4; ++mt)
#pragma unroll
    for (int nt = 0; nt < 2; ++nt) acc[mt][nt] = vzero;

  const int nk = K / TILE_K;
  int buf = 0;
  issue_tile(0, 0);  // 8 async copies per wave

  for (int kt = 0; kt < nk; ++kt) {
    if (kt + 1 < nk) {
      issue_tile((kt + 1) * TILE_K, buf ^ 1);
      WAIT_ASYNC(8);   // retire current tile; next tile stays in flight
    } else {
      WAIT_ASYNC(0);
    }
    __syncthreads();

    const bf16_t* sAhi = smem[buf][0];
    const bf16_t* sAlo = smem[buf][1];
    const bf16_t* sBhi = smem[buf][2];
    const bf16_t* sBlo = smem[buf][3];

    // A 16x32 bf16 frag: lanes 0-15 row=lr K{0..7,16..23}; lanes 16-31
    // K{8..15,24..31}.  B 32x16: lanes 0-15 col=lr K{0..7,16..23}; lanes 16-31 K{8..15,24..31}.
    v16bf aHi[4], aLo[4], bHi[2], bLo[2];
    const int ka = hw * 8;
#pragma unroll
    for (int mt = 0; mt < 4; ++mt) {
      int r = wr * 64 + mt * 16 + lr;
      aHi[mt] = cat8(*(v8bf*)(sAhi + r * LDSA + ka),
                     *(v8bf*)(sAhi + r * LDSA + ka + 16));
      aLo[mt] = cat8(*(v8bf*)(sAlo + r * LDSA + ka),
                     *(v8bf*)(sAlo + r * LDSA + ka + 16));
    }
    const int kb = hw * 8;
#pragma unroll
    for (int nt = 0; nt < 2; ++nt) {
      int r = wc * 32 + nt * 16 + lr;
      bHi[nt] = cat8(*(v8bf*)(sBhi + r * LDSA + kb),
                     *(v8bf*)(sBhi + r * LDSA + kb + 16));
      bLo[nt] = cat8(*(v8bf*)(sBlo + r * LDSA + kb),
                     *(v8bf*)(sBlo + r * LDSA + kb + 16));
    }

#pragma unroll
    for (int mt = 0; mt < 4; ++mt) {
#pragma unroll
      for (int nt = 0; nt < 2; ++nt) {
        acc[mt][nt] = __builtin_amdgcn_wmma_f32_16x16x32_bf16(
            false, aHi[mt], false, bHi[nt], (short)0, acc[mt][nt], false, false);
        acc[mt][nt] = __builtin_amdgcn_wmma_f32_16x16x32_bf16(
            false, aHi[mt], false, bLo[nt], (short)0, acc[mt][nt], false, false);
        acc[mt][nt] = __builtin_amdgcn_wmma_f32_16x16x32_bf16(
            false, aLo[mt], false, bHi[nt], (short)0, acc[mt][nt], false, false);
      }
    }
    __syncthreads();
    buf ^= 1;
  }

  // C/D layout: lanes0-15 N=lr, M=r; lanes16-31 N=lr, M=8+r.
#pragma unroll
  for (int mt = 0; mt < 4; ++mt) {
#pragma unroll
    for (int nt = 0; nt < 2; ++nt) {
      int n = n0 + wc * 32 + nt * 16 + lr;
      float bv = bias[n];
      int mbase = m0 + wr * 64 + mt * 16 + hw * 8;
      float* Cp = C + (size_t)mbase * N + n;
#pragma unroll
      for (int r = 0; r < 8; ++r)
        Cp[(size_t)r * N] = acc[mt][nt][r] + bv;
    }
  }
}

// ---------------- population: LN -> GELU -> exact top-k -> split ------------
#define POPN 2048

static __device__ __forceinline__ unsigned ordkey(float f) {
  unsigned u = __float_as_uint(f);
  return (u & 0x80000000u) ? ~u : (u | 0x80000000u);
}

__global__ __launch_bounds__(256) void cc_population(
    const float* __restrict__ hin, bf16_t* __restrict__ ahi,
    bf16_t* __restrict__ alo, const float* __restrict__ gamma,
    const float* __restrict__ beta, int k_keep) {
  __shared__ float sa[POPN];
  __shared__ float sred[256];
  __shared__ unsigned shist[256];
  __shared__ unsigned sscan[256];
  __shared__ unsigned s_prefix, s_kr;

  const int t = threadIdx.x;
  const size_t rowoff = (size_t)blockIdx.x * POPN;
  const float* hr = hin + rowoff;

  float sum = 0.f, ssq = 0.f;
  for (int i = t; i < POPN; i += 256) {
    float v = hr[i];
    sa[i] = v;
    sum += v;
    ssq += v * v;
  }
  sred[t] = sum;
  __syncthreads();
  for (int s = 128; s > 0; s >>= 1) {
    if (t < s) sred[t] += sred[t + s];
    __syncthreads();
  }
  float mu = sred[0] * (1.0f / POPN);
  __syncthreads();
  sred[t] = ssq;
  __syncthreads();
  for (int s = 128; s > 0; s >>= 1) {
    if (t < s) sred[t] += sred[t + s];
    __syncthreads();
  }
  float var = sred[0] * (1.0f / POPN) - mu * mu;
  float rinv = rsqrtf(var + 1e-5f);
  __syncthreads();

  for (int i = t; i < POPN; i += 256) {
    float hn = (sa[i] - mu) * rinv * gamma[i] + beta[i];
    sa[i] = 0.5f * hn * (1.0f + erff(hn * 0.70710678118654752f));
  }
  if (t == 0) {
    s_prefix = 0u;
    s_kr = (unsigned)k_keep;
  }
  __syncthreads();

  // 4-pass radix select for the k-th largest ordered key
  for (int pass = 0; pass < 4; ++pass) {
    shist[t] = 0u;
    __syncthreads();
    const int shift = 24 - 8 * pass;
    const unsigned pref = s_prefix;
    for (int i = t; i < POPN; i += 256) {
      unsigned u = ordkey(sa[i]);
      bool in;
      if (pass == 0) {
        in = true;
      } else {
        unsigned hs = (unsigned)(shift + 8);
        in = ((u >> hs) == (pref >> hs));
      }
      if (in) atomicAdd(&shist[(u >> shift) & 255u], 1u);
    }
    __syncthreads();
    if (t == 0) {
      unsigned kr = s_kr, pfx = s_prefix;
      for (int b = 255; b >= 0; --b) {
        unsigned c = shist[b];
        if (c >= kr) { pfx |= ((unsigned)b) << shift; break; }
        kr -= c;
      }
      s_prefix = pfx;
      s_kr = kr;
    }
    __syncthreads();
  }
  const unsigned T = s_prefix;
  const unsigned r = s_kr;

  // index-ordered rank among ties: thread t owns contiguous [8t, 8t+8)
  const int base_i = t * 8;
  unsigned cnt = 0;
#pragma unroll
  for (int j = 0; j < 8; ++j) cnt += (ordkey(sa[base_i + j]) == T) ? 1u : 0u;
  sscan[t] = cnt;
  __syncthreads();
  for (int off = 1; off < 256; off <<= 1) {
    unsigned add = (t >= off) ? sscan[t - off] : 0u;
    __syncthreads();
    sscan[t] += add;
    __syncthreads();
  }
  unsigned eqrank = sscan[t] - cnt;

#pragma unroll
  for (int j = 0; j < 8; ++j) {
    int i = base_i + j;
    float a = sa[i];
    unsigned u = ordkey(a);
    bool iseq = (u == T);
    bool keep = (u > T) || (iseq && (eqrank < r));
    if (iseq) ++eqrank;
    float v = keep ? a : 0.f;
    bf16_t hi = (bf16_t)v;
    ahi[rowoff + i] = hi;
    alo[rowoff + i] = (bf16_t)(v - (float)hi);
  }
}

// ---------------------------------------------------------------------------
extern "C" void kernel_launch(void* const* d_in, const int* in_sizes, int n_in,
                              void* d_out, int out_size, void* d_ws,
                              size_t ws_size, hipStream_t stream) {
  (void)in_sizes; (void)n_in; (void)out_size; (void)ws_size;
  const float* x        = (const float*)d_in[0];
  const float* W_in     = (const float*)d_in[1];
  const float* b_in     = (const float*)d_in[2];
  const float* ln_scale = (const float*)d_in[3];
  const float* ln_bias  = (const float*)d_in[4];
  const float* ff_w     = (const float*)d_in[5];
  const float* ff_b     = (const float*)d_in[6];
  const float* ff_mask  = (const float*)d_in[7];
  const float* W_out    = (const float*)d_in[8];
  const float* b_out    = (const float*)d_in[9];
  float* out = (float*)d_out;

  const int M = 16384, Din = 1024, N = 2048;
  const int kkeep = 204;  // max(1, int(0.1*2048))

  // workspace: activation hi/lo, then split weights
  bf16_t* Acthi = (bf16_t*)d_ws;
  const size_t nAct = (size_t)M * N;
  bf16_t* Actlo = Acthi + nAct;
  bf16_t* cur = Actlo + nAct;
  const size_t nWin = (size_t)N * Din;
  const size_t nFF  = (size_t)N * N;
  bf16_t* WinHi = cur; cur += nWin;
  bf16_t* WinLo = cur; cur += nWin;
  bf16_t* FfHi[3]; bf16_t* FfLo[3];
  for (int l = 0; l < 3; ++l) { FfHi[l] = cur; cur += nFF; FfLo[l] = cur; cur += nFF; }
  bf16_t* WoHi = cur; cur += nFF;
  bf16_t* WoLo = cur;

  // 1) split x and weights (FF weights masked) into bf16 hi/lo
  cc_prep_split<<<2048, 256, 0, stream>>>(x, nullptr, Acthi, Actlo, M * Din);
  cc_prep_split<<<512, 256, 0, stream>>>(W_in, nullptr, WinHi, WinLo, (int)nWin);
  for (int l = 0; l < 3; ++l)
    cc_prep_split<<<1024, 256, 0, stream>>>(ff_w + (size_t)l * nFF,
                                            ff_mask + (size_t)l * nFF,
                                            FfHi[l], FfLo[l], (int)nFF);
  cc_prep_split<<<1024, 256, 0, stream>>>(W_out, nullptr, WoHi, WoLo, (int)nFF);

  const dim3 gg(M / TILE_M, N / TILE_N);

  // 2) all GEMMs write f32 -> d_out; all populations re-split into ws
  cc_gemm_wmma<<<gg, 256, 0, stream>>>(Acthi, Actlo, WinHi, WinLo, b_in, out, M, N, Din);
  cc_population<<<M, 256, 0, stream>>>(out, Acthi, Actlo, ln_scale + 0 * N, ln_bias + 0 * N, kkeep);
  cc_gemm_wmma<<<gg, 256, 0, stream>>>(Acthi, Actlo, FfHi[0], FfLo[0], ff_b + 0 * N, out, M, N, N);
  cc_population<<<M, 256, 0, stream>>>(out, Acthi, Actlo, ln_scale + 1 * N, ln_bias + 1 * N, kkeep);
  cc_gemm_wmma<<<gg, 256, 0, stream>>>(Acthi, Actlo, FfHi[1], FfLo[1], ff_b + 1 * N, out, M, N, N);
  cc_population<<<M, 256, 0, stream>>>(out, Acthi, Actlo, ln_scale + 2 * N, ln_bias + 2 * N, kkeep);
  cc_gemm_wmma<<<gg, 256, 0, stream>>>(Acthi, Actlo, FfHi[2], FfLo[2], ff_b + 2 * N, out, M, N, N);
  cc_population<<<M, 256, 0, stream>>>(out, Acthi, Actlo, ln_scale + 3 * N, ln_bias + 3 * N, kkeep);
  cc_gemm_wmma<<<gg, 256, 0, stream>>>(Acthi, Actlo, WoHi, WoLo, b_out, out, M, N, N);
}
